// VQLayer_83743272337562
// MI455X (gfx1250) — matrix operand order repaired and checked by reading, not compiled, not run on hardware
//
#include <hip/hip_runtime.h>

// ---------------------------------------------------------------------------
// Types
// ---------------------------------------------------------------------------
typedef __attribute__((ext_vector_type(16))) _Float16 v16h;
typedef __attribute__((ext_vector_type(8)))  _Float16 v8h;
typedef __attribute__((ext_vector_type(8)))  float    v8f;
typedef __attribute__((ext_vector_type(4)))  float    f32x4;
typedef __attribute__((ext_vector_type(4)))  unsigned u32x4;
typedef __attribute__((ext_vector_type(8)))  int      i32x8;
typedef __attribute__((ext_vector_type(4)))  int      i32x4;

#define D_DIM   256
#define K_DIM   8192
#define N_ROWS  32768
#define CHUNK   64            // codebook columns per LDS chunk
#define ROWTILE 128           // rows per workgroup
#define LDS_ROW 528           // 512B of f16 data + 16B TDM pad (bank de-conflict)
#define OFF_A   0
#define OFF_B0  (ROWTILE * LDS_ROW)            // 67584
#define OFF_B1  (OFF_B0 + CHUNK * LDS_ROW)     // 101376
#define LDS_BYTES (OFF_B1 + CHUNK * LDS_ROW)   // 135168  (< 320KB/WGP)

// ---------------------------------------------------------------------------
// Tensor Data Mover: DMA a [tile_rows x 256] f16 tile (row stride 256 elems in
// global) into LDS at lds_off, padding +16B after every 512B row.
// Issued by ONE wave (EXEC ignored by TDM; descriptor values are uniform).
// ---------------------------------------------------------------------------
__device__ __forceinline__ void tdm_load_tile_f16(unsigned lds_off, void* lptr,
                                                  const void* gptr,
                                                  unsigned tile_rows,
                                                  unsigned tensor_rows) {
#if __has_builtin(__builtin_amdgcn_tensor_load_to_lds)
  unsigned long long ga = (unsigned long long)gptr;
  u32x4 g0;
  g0.x = 1u;                                   // count=1, user descriptor
  g0.y = lds_off;                              // lds_addr
  g0.z = (unsigned)ga;                         // global_addr[31:0]
  g0.w = (unsigned)(ga >> 32) | (2u << 30);    // global_addr[56:32] | type=2
  i32x8 g1;
  // data_size=2B (code1), pad_enable, pad_interval=128 dwords (code6),
  // pad_amount=4 dwords (code3), workgroup_mask=0 (not in cluster)
  g1[0] = (int)((1u << 16) | (1u << 20) | (6u << 22) | (3u << 25));
  g1[1] = (int)((256u & 0xFFFFu) << 16);                 // tensor_dim0 lo16 (<<16)
  g1[2] = (int)((256u >> 16) | (tensor_rows << 16));     // tensor_dim0 hi | tensor_dim1 lo
  g1[3] = (int)((tensor_rows >> 16) | (256u << 16));     // tensor_dim1 hi | tile_dim0=256
  g1[4] = (int)tile_rows;                                // tile_dim1 | tile_dim2=0
  g1[5] = 256;                                           // tensor_dim0_stride lo
  g1[6] = 0;                                             // stride0 hi | stride1 lo
  g1[7] = 0;                                             // stride1 hi
  i32x4 gz4 = {0, 0, 0, 0};
  i32x8 gz8 = {0, 0, 0, 0, 0, 0, 0, 0};
  __builtin_amdgcn_tensor_load_to_lds(g0, g1, gz4, gz4, gz8, 0);
#else
  // Fallback: synchronous copy by the issuing wave (32 lanes).
  const _Float16* g = (const _Float16*)gptr;
  int lane = threadIdx.x & 31;
  for (unsigned r = lane; r < tile_rows; r += 32) {
    v8h* drow = (v8h*)((char*)lptr + r * LDS_ROW);
    const v8h* srow = (const v8h*)(g + (unsigned long long)r * 256u);
    #pragma unroll
    for (int j = 0; j < 32; ++j) drow[j] = srow[j];
  }
#endif
  (void)lptr;
}

__device__ __forceinline__ void tdm_wait() {
#if __has_builtin(__builtin_amdgcn_s_wait_tensorcnt)
  __builtin_amdgcn_s_wait_tensorcnt(0);
#endif
}

// ---------------------------------------------------------------------------
// Kernel 1: embeddings [D,K] f32 -> ET [K,D] f16 (+ column norms), zero loss.
// One thread per code k; reads coalesced along K.
// ---------------------------------------------------------------------------
__global__ void vq_prep_embed(const float* __restrict__ E,
                              _Float16* __restrict__ ETh,
                              float* __restrict__ cnorm,
                              float* __restrict__ lossp) {
  int k = blockIdx.x * blockDim.x + threadIdx.x;
  if (k == 0) lossp[0] = 0.0f;
  if (k >= K_DIM) return;
  float nrm = 0.0f;
  for (int d = 0; d < D_DIM; d += 8) {
    v8h h;
    #pragma unroll
    for (int j = 0; j < 8; ++j) {
      float v = E[(unsigned long long)(d + j) * K_DIM + k];
      nrm = fmaf(v, v, nrm);
      h[j] = (_Float16)v;
    }
    *(v8h*)(ETh + (unsigned long long)k * D_DIM + d) = h;
  }
  cnorm[k] = nrm;
}

// ---------------------------------------------------------------------------
// Kernel 2: inputs f32 -> f16 (row-major, coalesced, 8 elems/thread).
// ---------------------------------------------------------------------------
__global__ void vq_prep_x(const float* __restrict__ X, _Float16* __restrict__ Xh) {
  unsigned long long i =
      ((unsigned long long)blockIdx.x * blockDim.x + threadIdx.x) * 8ull;
  f32x4 a = *(const f32x4*)(X + i);
  f32x4 b = *(const f32x4*)(X + i + 4);
  v8h h;
  #pragma unroll
  for (int j = 0; j < 4; ++j) { h[j] = (_Float16)a[j]; h[4 + j] = (_Float16)b[j]; }
  *(v8h*)(Xh + i) = h;
}

// ---------------------------------------------------------------------------
// Kernel 3: fused GEMM + argmin.  WG = 128 rows, 8 waves (4 rowgroups x 2
// colgroups); wave tile = 32 rows x 32 cols per chunk (2x2 WMMA tiles).
// A fragments persist in registers; B double-buffered via TDM.
// ---------------------------------------------------------------------------
__global__ __launch_bounds__(256, 1) void vq_argmin(
    const _Float16* __restrict__ Xh, const _Float16* __restrict__ ETh,
    const float* __restrict__ cnorm, unsigned* __restrict__ idxOut) {
  __shared__ unsigned char smem[LDS_BYTES];

  const int tid  = threadIdx.x;
  const int lane = tid & 31;
  const int wave = tid >> 5;
  const int rg   = wave >> 1;                 // 0..3 rowgroup
  const int cg   = wave & 1;                  // 0..1 colgroup
  const int n0   = blockIdx.x * ROWTILE;
  const unsigned ldsBase = (unsigned)(unsigned long long)(void*)smem;

  // Prologue: DMA A tile (128x256) and B chunk 0 (64x256), wave 0 only.
  if (wave == 0) {
    tdm_load_tile_f16(ldsBase + OFF_A, smem + OFF_A,
                      Xh + (unsigned long long)n0 * D_DIM, ROWTILE, N_ROWS);
    tdm_load_tile_f16(ldsBase + OFF_B0, smem + OFF_B0, ETh, CHUNK, K_DIM);
    tdm_wait();
  }
  __syncthreads();

  // A fragments -> registers (layout per ISA 16-bit A 16x32):
  // lane l: M=l&15; lanes<16 hold K {2v,2v+1} (v=0..3) and K+16 (v=4..7);
  // lanes>=16 same +8. Two 16B LDS loads per fragment.
  const int m    = lane & 15;
  const int ksel = lane >> 4;                 // 0/1
  v16h afrag[2][8];
  #pragma unroll
  for (int rt = 0; rt < 2; ++rt) {
    const unsigned char* arow =
        smem + OFF_A + (rg * 32 + rt * 16 + m) * LDS_ROW;
    #pragma unroll
    for (int kk = 0; kk < 8; ++kk) {
      const unsigned char* p = arow + (kk * 32 + ksel * 8) * 2;
      union { f32x4 q[2]; v16h h; } u;
      u.q[0] = *(const f32x4*)(p);
      u.q[1] = *(const f32x4*)(p + 32);       // K += 16
      afrag[rt][kk] = u.h;
    }
  }

  float    runv[2][8];
  unsigned runi[2][8];
  #pragma unroll
  for (int rt = 0; rt < 2; ++rt)
    #pragma unroll
    for (int v = 0; v < 8; ++v) { runv[rt][v] = 3.4e38f; runi[rt][v] = 0u; }

  const int nn = lane & 15;
  for (int c = 0; c < K_DIM / CHUNK; ++c) {
    const unsigned offB  = (c & 1) ? OFF_B1 : OFF_B0;
    const unsigned offBn = (c & 1) ? OFF_B0 : OFF_B1;
    if (wave == 0 && (c + 1) < K_DIM / CHUNK)
      tdm_load_tile_f16(ldsBase + offBn, smem + offBn,
                        ETh + (unsigned long long)(c + 1) * CHUNK * D_DIM,
                        CHUNK, K_DIM);

    v8f acc[2][2];
    #pragma unroll
    for (int rt = 0; rt < 2; ++rt)
      #pragma unroll
      for (int ct = 0; ct < 2; ++ct) acc[rt][ct] = (v8f)(0.0f);

    #pragma unroll
    for (int ct = 0; ct < 2; ++ct) {
      const unsigned char* brow =
          smem + offB + (cg * 32 + ct * 16 + nn) * LDS_ROW;
      #pragma unroll
      for (int kk = 0; kk < 8; ++kk) {
        // B layout (16-bit 32x16): lane l: N=l&15; K = (l>=16?8:0)+0..7 and +16
        union { f32x4 q[2]; v16h h; } ub;
        const unsigned char* p = brow + (kk * 32 + ksel * 8) * 2;
        ub.q[0] = *(const f32x4*)(p);
        ub.q[1] = *(const f32x4*)(p + 32);
        acc[0][ct] = __builtin_amdgcn_wmma_f32_16x16x32_f16(
            false, afrag[0][kk], false, ub.h, (short)0, acc[0][ct], false, false);
        acc[1][ct] = __builtin_amdgcn_wmma_f32_16x16x32_f16(
            false, afrag[1][kk], false, ub.h, (short)0, acc[1][ct], false, false);
      }
    }

    // score = ||e||^2 - 2*dot ; keep lane-local running argmin
    #pragma unroll
    for (int ct = 0; ct < 2; ++ct) {
      const unsigned col = (unsigned)(c * CHUNK + cg * 32 + ct * 16 + nn);
      const float cn = cnorm[col];
      #pragma unroll
      for (int rt = 0; rt < 2; ++rt)
        #pragma unroll
        for (int v = 0; v < 8; ++v) {
          float s = fmaf(acc[rt][ct][v], -2.0f, cn);
          if (s < runv[rt][v]) { runv[rt][v] = s; runi[rt][v] = col; }
        }
    }

    if (wave == 0) tdm_wait();
    __syncthreads();
  }

  // Cross-lane argmin within each 16-lane half (rows split per C layout).
  #pragma unroll
  for (int rt = 0; rt < 2; ++rt)
    #pragma unroll
    for (int v = 0; v < 8; ++v) {
      float x = runv[rt][v]; unsigned xi = runi[rt][v];
      #pragma unroll
      for (int mm = 1; mm < 16; mm <<= 1) {
        float    ov = __shfl_xor(x, mm, 32);
        unsigned oi = __shfl_xor(xi, mm, 32);
        if (ov < x || (ov == x && oi < xi)) { x = ov; xi = oi; }
      }
      runv[rt][v] = x; runi[rt][v] = xi;
    }

  // Cross-wave (cg pair) combine via LDS, then write indices.
  float*    redv = (float*)smem;
  unsigned* redi = (unsigned*)(smem + 512);
  if (cg == 0 && (lane & 15) == 0) {
    #pragma unroll
    for (int rt = 0; rt < 2; ++rt)
      #pragma unroll
      for (int v = 0; v < 8; ++v) {
        int lrow = rg * 32 + rt * 16 + v + ksel * 8;
        redv[lrow] = runv[rt][v];
        redi[lrow] = runi[rt][v];
      }
  }
  __syncthreads();
  if (cg == 1 && (lane & 15) == 0) {
    #pragma unroll
    for (int rt = 0; rt < 2; ++rt)
      #pragma unroll
      for (int v = 0; v < 8; ++v) {
        int lrow = rg * 32 + rt * 16 + v + ksel * 8;
        float a = redv[lrow]; unsigned ai = redi[lrow];
        float x = runv[rt][v]; unsigned xi = runi[rt][v];
        if (a < x || (a == x && ai < xi)) { x = a; xi = ai; }
        idxOut[n0 + lrow] = xi;
      }
  }
}

// ---------------------------------------------------------------------------
// Kernel 4: gather codes (f32, L2-resident) -> quantized_st; per-block loss
// partials (deterministic reduction).
// ---------------------------------------------------------------------------
__global__ void vq_gather(const float* __restrict__ X, const float* __restrict__ E,
                          const unsigned* __restrict__ idx,
                          float* __restrict__ out, float* __restrict__ partials) {
  __shared__ float wsum[8];
  const int lane = threadIdx.x & 31;
  const int wave = threadIdx.x >> 5;
  const int row  = blockIdx.x * 8 + wave;          // one wave per row
  const unsigned k = idx[row];
  float part = 0.0f;
  #pragma unroll
  for (int i = 0; i < 8; ++i) {
    int d = lane + i * 32;
    float e  = E[(unsigned long long)d * K_DIM + k];
    float xv = X[(unsigned long long)row * D_DIM + d];
    out[(unsigned long long)row * D_DIM + d] = e;  // quantized_st == quantized
    float df = e - xv;
    part = fmaf(df, df, part);
  }
  #pragma unroll
  for (int mm = 16; mm > 0; mm >>= 1) part += __shfl_xor(part, mm, 32);
  if (lane == 0) wsum[wave] = part;
  __syncthreads();
  if (threadIdx.x == 0) {
    float s = 0.0f;
    #pragma unroll
    for (int j = 0; j < 8; ++j) s += wsum[j];
    partials[blockIdx.x] = s;
  }
}

__global__ void vq_loss_final(const float* __restrict__ partials,
                              float* __restrict__ lossp) {
  __shared__ float red[256];
  float s = 0.0f;
  for (int i = threadIdx.x; i < 4096; i += 256) s += partials[i];
  red[threadIdx.x] = s;
  __syncthreads();
  for (int stride = 128; stride > 0; stride >>= 1) {
    if (threadIdx.x < stride) red[threadIdx.x] += red[threadIdx.x + stride];
    __syncthreads();
  }
  if (threadIdx.x == 0)
    lossp[0] = red[0] * (1.25f / 8388608.0f);  // (1 + 0.25) * mean
}

// ---------------------------------------------------------------------------
// Launch
// ---------------------------------------------------------------------------
extern "C" void kernel_launch(void* const* d_in, const int* in_sizes, int n_in,
                              void* d_out, int out_size, void* d_ws, size_t ws_size,
                              hipStream_t stream) {
  (void)in_sizes; (void)n_in; (void)out_size; (void)ws_size;
  const float* x   = (const float*)d_in[0];   // [64,512,256]
  const float* emb = (const float*)d_in[1];   // [256,8192]
  float* out = (float*)d_out;                 // 8388608 quantized + 1 loss

  char* ws = (char*)d_ws;
  _Float16* ETh   = (_Float16*)(ws);                       // 4,194,304 B
  float*    cnorm = (float*)(ws + 4194304);                //    32,768 B
  _Float16* Xh    = (_Float16*)(ws + 4227072);             // 16,777,216 B
  unsigned* idx   = (unsigned*)(ws + 21004288);            //   131,072 B
  float*    parts = (float*)(ws + 21135360);               //    16,384 B
  float*    lossp = out + 8388608;

  vq_prep_embed<<<K_DIM / 256, 256, 0, stream>>>(emb, ETh, cnorm, lossp);
  vq_prep_x<<<4096, 256, 0, stream>>>(x, Xh);
  vq_argmin<<<N_ROWS / ROWTILE, 256, 0, stream>>>(Xh, ETh, cnorm, idx);
  vq_gather<<<4096, 256, 0, stream>>>(x, emb, idx, out, parts);
  vq_loss_final<<<1, 256, 0, stream>>>(parts, lossp);
}
